// SemMol_71708773974600
// MI455X (gfx1250) — matrix operand order repaired and checked by reading, not compiled, not run on hardware
//
#include <hip/hip_runtime.h>
#include <cstddef>

// ---------------------------------------------------------------------------
// Sizes from the reference: B=128, L=1024, D=128, K=4096, M=5
// N = B*L = 131072 token rows.
// ---------------------------------------------------------------------------
#define BQ 128
#define LQ 1024
#define DQ 128
#define KQ 4096
#define MQ 5
#define TAUQ 0.07f
#define TTHR 0.5f

#define BLOCK_ROWS 32
#define XP 264      // bf16 LDS pitch (max 256 cols + 8 pad; row stride 528B, 16B-aligned)
#define HP 132      // f32 LDS pitch

typedef __bf16 bf16;
typedef __attribute__((ext_vector_type(16))) __bf16 v16bf;
typedef __attribute__((ext_vector_type(8)))  __bf16 v8bf;
typedef __attribute__((ext_vector_type(8)))  float  v8f;

// ---------------------------------------------------------------------------
// WMMA fragment loads (wave32, V_WMMA_F32_16X16X32_BF16 layouts per ISA 7.12.2)
//   A 16x32 bf16: lane l (l0=l&15,hi=l>>4) = row l0; elems 0..7 = K[hi*8..+8],
//                 elems 8..15 = K[16+hi*8..+8]  -> two 16B LDS loads.
//   B 32x16 bf16: lane = column n; elems 0..7 = K[hi*8..+8], elems 8..15 =
//                 K[16+hi*8..+8] in the transposed weight WT[n][k] -> two 16B global loads.
// ---------------------------------------------------------------------------
__device__ __forceinline__ v16bf load_afrag(const bf16* Xb, int mt, int kt, int l0, int hi) {
  const bf16* p = Xb + (mt * 16 + l0) * XP + kt * 32;
  v8bf c0 = *(const v8bf*)(p + hi * 8);
  v8bf c1 = *(const v8bf*)(p + 16 + hi * 8);
  v16bf a;
#pragma unroll
  for (int i = 0; i < 8; ++i) { a[i] = c0[i]; a[i + 8] = c1[i]; }
  return a;
}

__device__ __forceinline__ v16bf load_bfrag(const bf16* WT, int din, int kt, int nt, int l0, int hi) {
  const bf16* p = WT + (size_t)(nt * 16 + l0) * din + kt * 32;
  v8bf c0 = *(const v8bf*)(p + hi * 8);
  v8bf c1 = *(const v8bf*)(p + 16 + hi * 8);
  v16bf b;
#pragma unroll
  for (int i = 0; i < 8; ++i) { b[i] = c0[i]; b[i + 8] = c1[i]; }
  return b;
}

// One GEMM stage: H[32][DOUT=128] = act(Xb[32][DIN] @ W + bias).
// 8 waves, wave w owns column-tile nt=w and both row tiles (mt=0,1).
// Writes back to Xb (bf16, for chaining) or Hacc (f32, before LayerNorm).
template<int DIN, bool RELU, bool TO_LDS>
__device__ __forceinline__ void gemm_stage(const bf16* Xb, const bf16* WT, const float* bias,
                                           bf16* outLds, float* outF) {
  const int t = threadIdx.x, lane = t & 31, wave = t >> 5;
  const int l0 = lane & 15, hi = lane >> 4, nt = wave;
  v8f acc0 = {}, acc1 = {};
#pragma unroll
  for (int kt = 0; kt < DIN / 32; ++kt) {
    v16bf bf = load_bfrag(WT, DIN, kt, nt, l0, hi);
    v16bf a0 = load_afrag(Xb, 0, kt, l0, hi);
    v16bf a1 = load_afrag(Xb, 1, kt, l0, hi);
    acc0 = __builtin_amdgcn_wmma_f32_16x16x32_bf16(false, a0, false, bf, (short)0, acc0, false, false);
    acc1 = __builtin_amdgcn_wmma_f32_16x16x32_bf16(false, a1, false, bf, (short)0, acc1, false, false);
  }
  const float bv = bias[nt * 16 + l0];
  __syncthreads();  // all reads of Xb done before anyone overwrites it
  const int col = nt * 16 + l0;
#pragma unroll
  for (int v = 0; v < 8; ++v) {
    float x0 = acc0[v] + bv, x1 = acc1[v] + bv;
    if (RELU) { x0 = fmaxf(x0, 0.f); x1 = fmaxf(x1, 0.f); }
    const int r0 = hi * 8 + v, r1 = 16 + hi * 8 + v;  // C/D layout: vgpr v -> M=v(+8)
    if (TO_LDS) { outLds[r0 * XP + col] = (bf16)x0; outLds[r1 * XP + col] = (bf16)x1; }
    else        { outF[r0 * HP + col] = x0;          outF[r1 * HP + col] = x1; }
  }
  __syncthreads();
}

// Stage 16 consecutive f32 -> 16 bf16 in LDS (two aligned 16B ds stores).
__device__ __forceinline__ void stage16(const float* src, bf16* dst) {
  float tmp[16];
  *(float4*)(tmp + 0)  = ((const float4*)src)[0];
  *(float4*)(tmp + 4)  = ((const float4*)src)[1];
  *(float4*)(tmp + 8)  = ((const float4*)src)[2];
  *(float4*)(tmp + 12) = ((const float4*)src)[3];
  v8bf o0, o1;
#pragma unroll
  for (int i = 0; i < 8; ++i) { o0[i] = (bf16)tmp[i]; o1[i] = (bf16)tmp[8 + i]; }
  *(v8bf*)(dst) = o0;
  *(v8bf*)(dst + 8) = o1;
}

// ---------------------------------------------------------------------------
// Fused per-stream kernel: [optional pre-proj] -> W1+ReLU -> W2 -> LayerNorm
// [-> l2norm]. 32 rows per block; fuse stage concatenates h_pos (DIN1=256).
// Safe to run with out == X (block reads its rows before writing them).
// ---------------------------------------------------------------------------
template<bool HAS_PRE, int DIN1, bool DO_L2>
__global__ __launch_bounds__(256) void fused_stream_kernel(
    const float* X,
    const unsigned short* WpreT_u, const float* bpre,
    const unsigned short* W1T_u, const float* b1,
    const unsigned short* W2T_u, const float* b2,
    const float* g, const float* be,
    const float* hpos,
    float* out)
{
  __shared__ bf16 Xb[BLOCK_ROWS * XP];
  __shared__ float Hacc[BLOCK_ROWS * HP];
  __shared__ float red[BLOCK_ROWS * 8], red2[BLOCK_ROWS * 8];
  __shared__ float rstat[BLOCK_ROWS];

  const int t = threadIdx.x;
  const long rowbase = (long)blockIdx.x * BLOCK_ROWS;
  const int b = (int)(rowbase >> 10);  // rows/block divides L -> single b per block

  // Stage input rows as bf16 into LDS: thread t owns row r, cols sub*16..+16
  // -> 4x global_load_b128 + 2x ds_store_b128 (row pitch 528B keeps 16B align).
  const int r = t >> 3, sub = t & 7;
  stage16(X + (rowbase + r) * DQ + sub * 16, Xb + r * XP + sub * 16);
  if (DIN1 == 256) {  // concat broadcast h_pos[b] in cols 128..255 (L2-hot)
    stage16(hpos + b * DQ + sub * 16, Xb + r * XP + 128 + sub * 16);
  }
  __syncthreads();

  if (HAS_PRE)
    gemm_stage<128, false, true >(Xb, (const bf16*)WpreT_u, bpre, Xb, Hacc);
  gemm_stage<DIN1,  true, true >(Xb, (const bf16*)W1T_u,   b1,   Xb, Hacc);
  gemm_stage<128,  false, false>(Xb, (const bf16*)W2T_u,   b2,   Xb, Hacc);

  // LayerNorm (biased var, eps=1e-5); 8 threads per row, 16 elems each
  float s = 0.f, ss = 0.f;
#pragma unroll
  for (int i = 0; i < 16; ++i) {
    const float x = Hacc[r * HP + sub * 16 + i];
    s += x; ss += x * x;
  }
  red[r * 8 + sub] = s; red2[r * 8 + sub] = ss;
  __syncthreads();
  if (sub == 0) {
    float S = 0.f, SS = 0.f;
#pragma unroll
    for (int i = 0; i < 8; ++i) { S += red[r * 8 + i]; SS += red2[r * 8 + i]; }
    const float mu = S * (1.f / 128.f);
    const float var = SS * (1.f / 128.f) - mu * mu;
    red[r * 8] = mu;
    rstat[r] = rsqrtf(var + 1e-5f);
  }
  __syncthreads();
  const float mu = red[r * 8], rs = rstat[r];
  float yv[16]; float ysq = 0.f;
#pragma unroll
  for (int i = 0; i < 16; ++i) {
    const int c = sub * 16 + i;
    const float y = (Hacc[r * HP + c] - mu) * rs * g[c] + be[c];
    yv[i] = y; ysq += y * y;
  }
  if (DO_L2) {
    __syncthreads();
    red2[r * 8 + sub] = ysq;
    __syncthreads();
    float S = 0.f;
#pragma unroll
    for (int i = 0; i < 8; ++i) S += red2[r * 8 + i];
    const float inv = 1.f / fmaxf(sqrtf(S), 1e-12f);
#pragma unroll
    for (int i = 0; i < 16; ++i) yv[i] *= inv;
  }
  float4* dst = (float4*)(out + (rowbase + r) * DQ + sub * 16);
#pragma unroll
  for (int i = 0; i < 4; ++i)
    dst[i] = make_float4(yv[4 * i], yv[4 * i + 1], yv[4 * i + 2], yv[4 * i + 3]);
}

// ---------------------------------------------------------------------------
// Weight prep: f32 W[din][dout] -> bf16 W^T[dout][din]
// ---------------------------------------------------------------------------
__global__ void wt_kernel(const float* W, unsigned short* WTu, int din, int dout) {
  const int i = blockIdx.x * blockDim.x + threadIdx.x;
  if (i >= din * dout) return;
  const int o = i % dout, r = i / dout;
  ((bf16*)WTu)[(size_t)o * din + r] = (bf16)W[i];
}

// h_anchor = mean_L(h1); a = l2norm(h_anchor).  One block per b, 128 threads.
__global__ void anchor_kernel(const float* h1, float* a) {
  const int b = blockIdx.x, d = threadIdx.x;
  const float* p = h1 + (size_t)b * LQ * DQ + d;
  float s = 0.f;
  for (int l = 0; l < LQ; ++l) s += p[(size_t)l * DQ];
  s *= (1.f / (float)LQ);
  __shared__ float buf[DQ];
  __shared__ float inv;
  buf[d] = s * s; __syncthreads();
  for (int off = 64; off > 0; off >>= 1) { if (d < off) buf[d] += buf[d + off]; __syncthreads(); }
  if (d == 0) inv = 1.f / fmaxf(sqrtf(buf[0]), 1e-12f);
  __syncthreads();
  a[b * DQ + d] = s * inv;
}

__global__ void cnorm_kernel(const float* C, float* cn) {
  const int k = blockIdx.x * blockDim.x + threadIdx.x;
  if (k >= KQ) return;
  float s = 0.f;
  for (int d = 0; d < DQ; ++d) { const float x = C[(size_t)k * DQ + d]; s += x * x; }
  cn[k] = fmaxf(sqrtf(s), 1e-12f);
}

// sim[b][k] = <a[b], C[k]> / ||C[k]||.  grid (K/256, B).
__global__ void sim_kernel(const float* a, const float* C, const float* cn, float* sim) {
  const int b = blockIdx.y;
  const int k = blockIdx.x * 256 + threadIdx.x;
  __shared__ float av[DQ];
  if (threadIdx.x < DQ) av[threadIdx.x] = a[b * DQ + threadIdx.x];
  __syncthreads();
  float s = 0.f;
  const float* c = C + (size_t)k * DQ;
  for (int d = 0; d < DQ; ++d) s += av[d] * c[d];
  sim[(size_t)b * KQ + k] = s / cn[k];
}

// Top-5 + softmax(./TAU) + weighted codebook rows. Selected sim entries are
// overwritten with -1e30, which also realizes the ~in_top mask for the loss.
__global__ void toppos_kernel(float* sim, const float* C, float* pos) {
  const int b = blockIdx.x, t = threadIdx.x;
  __shared__ float vmax[256]; __shared__ int imax[256];
  __shared__ float topv[MQ]; __shared__ int topi[MQ]; __shared__ float w[MQ];
  float* srow = sim + (size_t)b * KQ;
  for (int m = 0; m < MQ; ++m) {
    float bv = -1e30f; int bi = 0;
    for (int k = t; k < KQ; k += 256) { const float v = srow[k]; if (v > bv) { bv = v; bi = k; } }
    vmax[t] = bv; imax[t] = bi; __syncthreads();
    for (int off = 128; off > 0; off >>= 1) {
      if (t < off && vmax[t + off] > vmax[t]) { vmax[t] = vmax[t + off]; imax[t] = imax[t + off]; }
      __syncthreads();
    }
    if (t == 0) { topv[m] = vmax[0]; topi[m] = imax[0]; srow[imax[0]] = -1e30f; }
    __syncthreads();
  }
  if (t == 0) {
    float mx = topv[0];
    for (int m = 1; m < MQ; ++m) mx = fmaxf(mx, topv[m]);
    float s = 0.f;
    for (int m = 0; m < MQ; ++m) { w[m] = __expf((topv[m] - mx) / TAUQ); s += w[m]; }
    for (int m = 0; m < MQ; ++m) w[m] /= s;
  }
  __syncthreads();
  if (t < DQ) {
    float p = 0.f;
    for (int m = 0; m < MQ; ++m) p += w[m] * C[(size_t)topi[m] * DQ + t];
    pos[b * DQ + t] = p;
  }
}

// h_pos = pos2+pos3; pos_logit = <a, l2norm(h_pos)>/TAU
__global__ void posfin_kernel(const float* pos2, const float* pos3, const float* a,
                              float* hpos, float* pos_logit) {
  const int b = blockIdx.x, d = threadIdx.x;
  const float hp = pos2[b * DQ + d] + pos3[b * DQ + d];
  __shared__ float buf[DQ];
  __shared__ float inv;
  buf[d] = hp * hp; __syncthreads();
  for (int off = 64; off > 0; off >>= 1) { if (d < off) buf[d] += buf[d + off]; __syncthreads(); }
  if (d == 0) inv = 1.f / fmaxf(sqrtf(buf[0]), 1e-12f);
  __syncthreads();
  hpos[b * DQ + d] = hp;
  buf[d] = a[b * DQ + d] * hp * inv;
  __syncthreads();
  for (int off = 64; off > 0; off >>= 1) { if (d < off) buf[d] += buf[d + off]; __syncthreads(); }
  if (d == 0) pos_logit[b] = buf[0] / TAUQ;
}

// per-sample InfoNCE: logsumexp([pos_logit, kept sim/TAU]) - pos_logit.
// Top-M entries were already set to -1e30 so the sim>T_THR test masks them.
__global__ void loss_kernel(const float* sim, const float* pos_logit, float* per) {
  const int b = blockIdx.x, t = threadIdx.x;
  const float* srow = sim + (size_t)b * KQ;
  const float pl = pos_logit[b];
  float mx = pl;
  for (int k = t; k < KQ; k += 256) { const float v = srow[k]; if (v > TTHR) mx = fmaxf(mx, v / TAUQ); }
  __shared__ float buf[256];
  buf[t] = mx; __syncthreads();
  for (int off = 128; off > 0; off >>= 1) { if (t < off) buf[t] = fmaxf(buf[t], buf[t + off]); __syncthreads(); }
  mx = buf[0]; __syncthreads();
  float s = 0.f;
  for (int k = t; k < KQ; k += 256) { const float v = srow[k]; if (v > TTHR) s += __expf(v / TAUQ - mx); }
  buf[t] = s; __syncthreads();
  for (int off = 128; off > 0; off >>= 1) { if (t < off) buf[t] += buf[t + off]; __syncthreads(); }
  if (t == 0) per[b] = logf(buf[0] + __expf(pl - mx)) + mx - pl;
}

__global__ void lossfin_kernel(const float* per2, const float* per3, float* out) {
  const int t = threadIdx.x;
  __shared__ float buf[BQ];
  buf[t] = per2[t] + per3[t];
  __syncthreads();
  for (int off = 64; off > 0; off >>= 1) { if (t < off) buf[t] += buf[t + off]; __syncthreads(); }
  if (t == 0) out[0] = buf[0] / (float)BQ;
}

// ---------------------------------------------------------------------------
extern "C" void kernel_launch(void* const* d_in, const int* in_sizes, int n_in,
                              void* d_out, int out_size, void* d_ws, size_t ws_size,
                              hipStream_t stream) {
  (void)in_sizes; (void)n_in; (void)out_size; (void)ws_size;
  const float* z1 = (const float*)d_in[0];
  const float* z2 = (const float*)d_in[1];
  const float* z3 = (const float*)d_in[2];
  const float* C2 = (const float*)d_in[3];
  const float* C3 = (const float*)d_in[4];
  const float* mlp_W1 = (const float*)d_in[5];
  const float* mlp_b1 = (const float*)d_in[6];
  const float* mlp_W2 = (const float*)d_in[7];
  const float* mlp_b2 = (const float*)d_in[8];
  const float* mlp_g  = (const float*)d_in[9];
  const float* mlp_be = (const float*)d_in[10];
  const float* Wb_W = (const float*)d_in[11];
  const float* Wb_b = (const float*)d_in[12];
  const float* Wc_W = (const float*)d_in[13];
  const float* Wc_b = (const float*)d_in[14];
  const float* f_W1[3] = {(const float*)d_in[15], (const float*)d_in[21], (const float*)d_in[27]};
  const float* f_b1[3] = {(const float*)d_in[16], (const float*)d_in[22], (const float*)d_in[28]};
  const float* f_W2[3] = {(const float*)d_in[17], (const float*)d_in[23], (const float*)d_in[29]};
  const float* f_b2[3] = {(const float*)d_in[18], (const float*)d_in[24], (const float*)d_in[30]};
  const float* f_g [3] = {(const float*)d_in[19], (const float*)d_in[25], (const float*)d_in[31]};
  const float* f_be[3] = {(const float*)d_in[20], (const float*)d_in[26], (const float*)d_in[32]};

  char* wsp = (char*)d_ws;
  auto alloc = [&](size_t bytes) { char* p = wsp; wsp += (bytes + 255) & ~(size_t)255; return p; };
  unsigned short* WT_m1 = (unsigned short*)alloc(128 * 128 * 2);
  unsigned short* WT_m2 = (unsigned short*)alloc(128 * 128 * 2);
  unsigned short* WT_b  = (unsigned short*)alloc(128 * 128 * 2);
  unsigned short* WT_c  = (unsigned short*)alloc(128 * 128 * 2);
  unsigned short* WT_fw1[3]; unsigned short* WT_fw2[3];
  for (int s = 0; s < 3; ++s) {
    WT_fw1[s] = (unsigned short*)alloc(256 * 128 * 2);
    WT_fw2[s] = (unsigned short*)alloc(128 * 128 * 2);
  }
  float* a_vec = (float*)alloc((size_t)BQ * DQ * 4);
  float* cn2   = (float*)alloc((size_t)KQ * 4);
  float* cn3   = (float*)alloc((size_t)KQ * 4);
  float* sim2  = (float*)alloc((size_t)BQ * KQ * 4);
  float* sim3  = (float*)alloc((size_t)BQ * KQ * 4);
  float* pos2  = (float*)alloc((size_t)BQ * DQ * 4);
  float* pos3  = (float*)alloc((size_t)BQ * DQ * 4);
  float* hposv = (float*)alloc((size_t)BQ * DQ * 4);
  float* plog  = (float*)alloc((size_t)BQ * 4);
  float* per2  = (float*)alloc((size_t)BQ * 4);
  float* per3  = (float*)alloc((size_t)BQ * 4);

  const size_t NT = (size_t)BQ * LQ * DQ;  // 16,777,216
  float* h1 = (float*)d_out;
  float* h2 = h1 + NT;
  float* h3 = h2 + NT;
  float* lossp = h1 + 3 * NT;

  // 1) weight prep (bf16, transposed; stays hot in 192MB L2)
  wt_kernel<<<64, 256, 0, stream>>>(mlp_W1, WT_m1, 128, 128);
  wt_kernel<<<64, 256, 0, stream>>>(mlp_W2, WT_m2, 128, 128);
  wt_kernel<<<64, 256, 0, stream>>>(Wb_W,   WT_b,  128, 128);
  wt_kernel<<<64, 256, 0, stream>>>(Wc_W,   WT_c,  128, 128);
  for (int s = 0; s < 3; ++s) {
    wt_kernel<<<128, 256, 0, stream>>>(f_W1[s], WT_fw1[s], 256, 128);
    wt_kernel<<<64,  256, 0, stream>>>(f_W2[s], WT_fw2[s], 128, 128);
  }

  // 2) stream MLPs -> h1,h2,h3 (staged into d_out, refined in place later)
  const dim3 gfull((unsigned)(BQ * LQ / BLOCK_ROWS));  // 4096 blocks
  fused_stream_kernel<false, 128, false><<<gfull, 256, 0, stream>>>(
      z1, nullptr, nullptr, WT_m1, mlp_b1, WT_m2, mlp_b2, mlp_g, mlp_be, nullptr, h1);
  fused_stream_kernel<true, 128, false><<<gfull, 256, 0, stream>>>(
      z2, WT_b, Wb_b, WT_m1, mlp_b1, WT_m2, mlp_b2, mlp_g, mlp_be, nullptr, h2);
  fused_stream_kernel<true, 128, false><<<gfull, 256, 0, stream>>>(
      z3, WT_c, Wc_b, WT_m1, mlp_b1, WT_m2, mlp_b2, mlp_g, mlp_be, nullptr, h3);

  // 3) anchor / similarity / top-k / loss (small)
  anchor_kernel<<<BQ, DQ, 0, stream>>>(h1, a_vec);
  cnorm_kernel<<<KQ / 256, 256, 0, stream>>>(C2, cn2);
  cnorm_kernel<<<KQ / 256, 256, 0, stream>>>(C3, cn3);
  sim_kernel<<<dim3(KQ / 256, BQ), 256, 0, stream>>>(a_vec, C2, cn2, sim2);
  sim_kernel<<<dim3(KQ / 256, BQ), 256, 0, stream>>>(a_vec, C3, cn3, sim3);
  toppos_kernel<<<BQ, 256, 0, stream>>>(sim2, C2, pos2);
  toppos_kernel<<<BQ, 256, 0, stream>>>(sim3, C3, pos3);
  posfin_kernel<<<BQ, DQ, 0, stream>>>(pos2, pos3, a_vec, hposv, plog);
  loss_kernel<<<BQ, 256, 0, stream>>>(sim2, plog, per2);
  loss_kernel<<<BQ, 256, 0, stream>>>(sim3, plog, per3);
  lossfin_kernel<<<1, BQ, 0, stream>>>(per2, per3, lossp);

  // 4) fuse stage, in place on d_out: concat(h, h_pos) -> MLP -> LN -> l2norm
  fused_stream_kernel<false, 256, true><<<gfull, 256, 0, stream>>>(
      h1, nullptr, nullptr, WT_fw1[0], f_b1[0], WT_fw2[0], f_b2[0], f_g[0], f_be[0], hposv, h1);
  fused_stream_kernel<false, 256, true><<<gfull, 256, 0, stream>>>(
      h2, nullptr, nullptr, WT_fw1[1], f_b1[1], WT_fw2[1], f_b2[1], f_g[1], f_be[1], hposv, h2);
  fused_stream_kernel<false, 256, true><<<gfull, 256, 0, stream>>>(
      h3, nullptr, nullptr, WT_fw1[2], f_b1[2], WT_fw2[2], f_b2[2], f_g[2], f_be[2], hposv, h3);
}
